// _Attention3d_deform_1228360646683
// MI455X (gfx1250) — matrix operand order repaired and checked by reading, not compiled, not run on hardware
//
#include <hip/hip_runtime.h>
#include <hip/hip_bf16.h>
#include <math.h>

typedef __bf16 bf16_t;
typedef __attribute__((ext_vector_type(16))) __bf16 bf16x16;
typedef __attribute__((ext_vector_type(8)))  __bf16 bf16x8;
typedef __attribute__((ext_vector_type(8)))  float  f32x8;

#define NTOT 32768   // H*W*D = 32^3
#define CCH  64

// ---------------------------------------------------------------------------
// WMMA: D(16x16 f32) += A(16x32 bf16) * B(32x16 bf16)
// A lane layout: m=L&15, half=L>>4; elems = two contiguous 8-elt runs at
//   k = half*8 .. +7  and  k = 16+half*8 .. +7   (within a 32-wide K chunk)
// B lane layout: n=L&15, half=L>>4; elems = two contiguous 8-elt runs at
//   k = half*8 .. +7  and  k = 16+half*8 .. +7   (same as A)
// ---------------------------------------------------------------------------
__device__ __forceinline__ f32x8 wmma_bf16(bf16x16 a, bf16x16 b, f32x8 c) {
  return __builtin_amdgcn_wmma_f32_16x16x32_bf16(false, a, false, b, (short)0, c,
                                                 false, false);
}

__device__ __forceinline__ bf16x16 frag_a(const bf16_t* row, int cb, int half) {
  bf16x8 lo = *(const bf16x8*)(row + cb + half * 8);
  bf16x8 hi = *(const bf16x8*)(row + cb + 16 + half * 8);
  return __builtin_shufflevector(lo, hi, 0, 1, 2, 3, 4, 5, 6, 7,
                                 8, 9, 10, 11, 12, 13, 14, 15);
}

__device__ __forceinline__ bf16x16 frag_b(const bf16_t* col, int cb, int half) {
  bf16x8 lo = *(const bf16x8*)(col + cb + half * 8);
  bf16x8 hi = *(const bf16x8*)(col + cb + 16 + half * 8);
  return __builtin_shufflevector(lo, hi, 0, 1, 2, 3, 4, 5, 6, 7,
                                 8, 9, 10, 11, 12, 13, 14, 15);
}

// gfx1250 async copy global->LDS (ASYNCcnt path), 16B per lane.
__device__ __forceinline__ void lds_async_b128(unsigned lds_off, const void* src) {
  asm volatile("global_load_async_to_lds_b128 %0, %1, off"
               : : "v"(lds_off), "v"(src) : "memory");
}
__device__ __forceinline__ void async_wait0() {
  asm volatile("s_wait_asynccnt 0x0" : : : "memory");
}

// ---------------------------------------------------------------------------
// K0a: pack weights to bf16 / WMMA-friendly layouts.
//  w1/w2/w3: [64][64] bf16 (proj1, conv1, proj2)
//  offwp: [27][96][64] bf16, o>=81 zero-padded   (A operand, offset conv)
//  dcnwp: [27][64][64] bf16                      (A operand, DCN GEMM)
// ---------------------------------------------------------------------------
__global__ void k_pack(const float* __restrict__ p1, const float* __restrict__ c1,
                       const float* __restrict__ p2, const float* __restrict__ offw,
                       const float* __restrict__ dcnw, bf16_t* __restrict__ w1,
                       bf16_t* __restrict__ w2, bf16_t* __restrict__ w3,
                       bf16_t* __restrict__ offwp, bf16_t* __restrict__ dcnwp) {
  const int i = blockIdx.x * 256 + threadIdx.x;
  if (i < 4096) {
    w1[i] = (bf16_t)p1[i];
    w2[i] = (bf16_t)c1[i];
    w3[i] = (bf16_t)p2[i];
  }
  if (i < 27 * 96 * 64) {
    const int c = i & 63, o = (i >> 6) % 96, t = i / (96 * 64);
    offwp[i] = (o < 81) ? (bf16_t)offw[(o * 64 + c) * 27 + t] : (bf16_t)0.0f;
  }
  if (i < 27 * 64 * 64) {
    const int c = i & 63, o = (i >> 6) & 63, t = i >> 12;
    dcnwp[i] = (bf16_t)dcnw[(o * 64 + c) * 27 + t];
  }
}

// K0b: f32 -> bf16 cast of x (x is already channel-last [N][64]).
__global__ void k_cast_bf16(const float* __restrict__ in, bf16_t* __restrict__ out) {
  const int i = (blockIdx.x * 256 + threadIdx.x) * 8;
  const float4 v0 = *(const float4*)(in + i);
  const float4 v1 = *(const float4*)(in + i + 4);
  bf16x8 o;
  o[0] = (bf16_t)v0.x; o[1] = (bf16_t)v0.y; o[2] = (bf16_t)v0.z; o[3] = (bf16_t)v0.w;
  o[4] = (bf16_t)v1.x; o[5] = (bf16_t)v1.y; o[6] = (bf16_t)v1.z; o[7] = (bf16_t)v1.w;
  *(bf16x8*)(out + i) = o;
}

// ---------------------------------------------------------------------------
// K1: proj1 + exact GELU.  256 thr = 8 waves -> 64o x 32n per block.
// A: Wl (LDS, async-staged).  B: xnl channel-last [N][64] bf16 (global).
// Out: hbuf f32 channel-major [64][N]  (depthwise input; also "u").
// ---------------------------------------------------------------------------
__global__ void k_proj1_gelu(const bf16_t* __restrict__ Wbf, const float* __restrict__ bias,
                             const bf16_t* __restrict__ Xnl, float* __restrict__ out) {
  __shared__ __align__(16) bf16_t Wl[64 * 64];
  const int tid = threadIdx.x;
  {
    const unsigned lbase = (unsigned)(uintptr_t)&Wl[0];
    const char* g = (const char*)Wbf;
    lds_async_b128(lbase + tid * 16u, g + tid * 16);
    lds_async_b128(lbase + 4096u + tid * 16u, g + 4096 + tid * 16);
    async_wait0();
  }
  __syncthreads();
  const int wave = tid >> 5, lane = tid & 31;
  const int ot = wave & 3, nt = wave >> 2;
  const int o0 = ot * 16;
  const int m = lane & 15, half = lane >> 4, nn = lane & 15;
  const int n = blockIdx.x * 32 + nt * 16 + nn;
  const bf16_t* arow = Wl + (o0 + m) * 64;
  const bf16_t* bcol = Xnl + (size_t)n * 64;
  f32x8 d = {0.f, 0.f, 0.f, 0.f, 0.f, 0.f, 0.f, 0.f};
#pragma unroll
  for (int cb = 0; cb < 64; cb += 32)
    d = wmma_bf16(frag_a(arow, cb, half), frag_b(bcol, cb, half), d);
#pragma unroll
  for (int r = 0; r < 8; ++r) {
    const int o = o0 + r + half * 8;
    float v = d[r] + bias[o];
    v = 0.5f * v * (1.0f + erff(v * 0.70710678118654752f));  // exact GELU
    out[(size_t)o * NTOT + n] = v;
  }
}

// ---------------------------------------------------------------------------
// K2: depthwise 5x5x5, pad 2.  grid (N/256, 64ch), channel-major f32.
// ---------------------------------------------------------------------------
__global__ void k_dw5(const float* __restrict__ in, const float* __restrict__ w,
                      const float* __restrict__ b, float* __restrict__ out) {
  const int n = blockIdx.x * blockDim.x + threadIdx.x;
  const int c = blockIdx.y;
  const int d = n & 31, wy = (n >> 5) & 31, h = n >> 10;
  const float* wc = w + c * 125;
  const float* ic = in + (size_t)c * NTOT;
  float acc = b[c];
  for (int kh = 0; kh < 5; ++kh) {
    const int hh = h + kh - 2;
    if ((unsigned)hh >= 32u) continue;
    for (int kw = 0; kw < 5; ++kw) {
      const int ww = wy + kw - 2;
      if ((unsigned)ww >= 32u) continue;
#pragma unroll
      for (int kd = 0; kd < 5; ++kd) {
        const int dd = d + kd - 2;
        if ((unsigned)dd >= 32u) continue;
        acc += wc[(kh * 5 + kw) * 5 + kd] * ic[((hh << 5) + ww) * 32 + dd];
      }
    }
  }
  out[(size_t)c * NTOT + n] = acc;
}

// ---------------------------------------------------------------------------
// K3: depthwise 7x7x7 dilation 3, pad 9 -> bf16 channel-last [N][64]
// (contiguous 128B rows: B operand for offset conv AND gather rows for DCN).
// ---------------------------------------------------------------------------
__global__ void k_dw7d3(const float* __restrict__ in, const float* __restrict__ w,
                        const float* __restrict__ b, bf16_t* __restrict__ out_nl) {
  const int n = blockIdx.x * blockDim.x + threadIdx.x;
  const int c = blockIdx.y;
  const int d = n & 31, wy = (n >> 5) & 31, h = n >> 10;
  const float* wc = w + c * 343;
  const float* ic = in + (size_t)c * NTOT;
  float acc = b[c];
  for (int kh = 0; kh < 7; ++kh) {
    const int hh = h + kh * 3 - 9;
    if ((unsigned)hh >= 32u) continue;
    for (int kw = 0; kw < 7; ++kw) {
      const int ww = wy + kw * 3 - 9;
      if ((unsigned)ww >= 32u) continue;
#pragma unroll
      for (int kd = 0; kd < 7; ++kd) {
        const int dd = d + kd * 3 - 9;
        if ((unsigned)dd >= 32u) continue;
        acc += wc[(kh * 7 + kw) * 7 + kd] * ic[((hh << 5) + ww) * 32 + dd];
      }
    }
  }
  out_nl[(size_t)n * 64 + c] = (bf16_t)acc;
}

// ---------------------------------------------------------------------------
// K4: offset conv (3x3x3 dense, 64->81ch) via WMMA, K = 27 taps x 64 ch.
// Block 192 thr = 6 waves; wave -> 16-o tile (81 padded to 96), 16 n.
// A: offwp [27][96][64] (contiguous).  B: a1nl rows (contiguous, zero if OOB).
// ---------------------------------------------------------------------------
__global__ void k_offconv(const bf16_t* __restrict__ offwp, const float* __restrict__ offb,
                          const bf16_t* __restrict__ a_nl, float* __restrict__ off_nl) {
  const int wave = threadIdx.x >> 5, lane = threadIdx.x & 31;
  const int o0 = wave * 16;
  const int m = lane & 15, half = lane >> 4, nn = lane & 15;
  const int n = blockIdx.x * 16 + nn;
  const int h = n >> 10, wy = (n >> 5) & 31, d = n & 31;

  f32x8 acc = {0.f, 0.f, 0.f, 0.f, 0.f, 0.f, 0.f, 0.f};
  for (int t = 0; t < 27; ++t) {
    const int hh = h + t / 9 - 1, ww = wy + (t / 3) % 3 - 1, dd = d + t % 3 - 1;
    const bool valid = ((unsigned)hh < 32u) && ((unsigned)ww < 32u) && ((unsigned)dd < 32u);
    const int flat = ((hh << 5) + ww) * 32 + dd;
    const bf16_t* arow = offwp + (size_t)(t * 96 + o0 + m) * 64;
    const bf16_t* brow = a_nl + (size_t)flat * 64;
    if (t < 26) __builtin_prefetch(offwp + (size_t)((t + 1) * 96 + o0 + m) * 64, 0, 1);
#pragma unroll
    for (int cb = 0; cb < 64; cb += 32) {
      bf16x16 b;
      if (valid) {
        b = frag_b(brow, cb, half);
      } else {
#pragma unroll
        for (int j = 0; j < 16; ++j) b[j] = (bf16_t)0.0f;
      }
      acc = wmma_bf16(frag_a(arow, cb, half), b, acc);
    }
  }
#pragma unroll
  for (int r = 0; r < 8; ++r) {
    const int o = o0 + r + half * 8;
    if (o < 81) off_nl[(size_t)n * 81 + o] = acc[r] + offb[o];
  }
}

// ---------------------------------------------------------------------------
// K5: deformable trilinear sampling -> LDS [27][16n][64c] bf16 -> DCN GEMM.
// Block 128 thr, 16 positions.  Phase1: 8 workers/pos gather 27 taps.
// Phase2: 4 waves x 16-o WMMA tiles, K = 1728 (27 taps x 64 ch).
// Out: dcn_nl [N][64] bf16 channel-last (conv1 B operand).
// ---------------------------------------------------------------------------
__global__ void __launch_bounds__(128)
k_deform_dcn(const float* __restrict__ off_nl, const bf16_t* __restrict__ a_nl,
             const bf16_t* __restrict__ dcnwp, const float* __restrict__ dcnb,
             bf16_t* __restrict__ dcn_nl) {
  __shared__ __align__(32) bf16_t smp[27][16][64];  // 55296 B
  const int n0 = blockIdx.x * 16;
  const int tid = threadIdx.x;
  const int nl = tid & 15;
  const int worker = tid >> 4;  // 0..7
  const int n = n0 + nl;
  const int h = n >> 10, wy = (n >> 5) & 31, d = n & 31;

  for (int t = worker; t < 27; t += 8) {
    const float oh = off_nl[(size_t)n * 81 + t * 3 + 0];
    const float ow = off_nl[(size_t)n * 81 + t * 3 + 1];
    const float od = off_nl[(size_t)n * 81 + t * 3 + 2];
    const float ph = (float)(h + t / 9 - 1) + oh;
    const float pw = (float)(wy + (t / 3) % 3 - 1) + ow;
    const float pd = (float)(d + t % 3 - 1) + od;
    const float fh = floorf(ph), fw = floorf(pw), fd = floorf(pd);
    const int ih = (int)fh, iw = (int)fw, id = (int)fd;
    const float rh = ph - fh, rw = pw - fw, rd = pd - fd;
    float wt8[8];
    int   fl8[8];
#pragma unroll
    for (int cr = 0; cr < 8; ++cr) {
      const int ch = (cr >> 2) & 1, cw = (cr >> 1) & 1, cd = cr & 1;
      const int ah = ih + ch, aw = iw + cw, ad = id + cd;
      const bool valid = ((unsigned)ah < 32u) && ((unsigned)aw < 32u) && ((unsigned)ad < 32u);
      const float wgt = (ch ? rh : 1.f - rh) * (cw ? rw : 1.f - rw) * (cd ? rd : 1.f - rd);
      wt8[cr] = valid ? wgt : 0.f;
      fl8[cr] = valid ? ((ah * 32 + aw) * 32 + ad) : 0;
    }
#pragma unroll
    for (int cc = 0; cc < 4; ++cc) {  // 4 chunks of 16 channels
      float acc[16];
#pragma unroll
      for (int c = 0; c < 16; ++c) acc[c] = 0.f;
#pragma unroll
      for (int cr = 0; cr < 8; ++cr) {
        const float wgt = wt8[cr];
        const bf16x16 rv = *(const bf16x16*)(a_nl + (size_t)fl8[cr] * 64 + cc * 16);
#pragma unroll
        for (int c = 0; c < 16; ++c) acc[c] += wgt * (float)rv[c];
      }
      bf16x8 s0, s1;
#pragma unroll
      for (int c = 0; c < 8; ++c) { s0[c] = (bf16_t)acc[c]; s1[c] = (bf16_t)acc[c + 8]; }
      *(bf16x8*)(&smp[t][nl][cc * 16 + 0]) = s0;
      *(bf16x8*)(&smp[t][nl][cc * 16 + 8]) = s1;
    }
  }
  __syncthreads();

  const int wave = tid >> 5, lane = tid & 31;
  const int o0 = wave * 16;
  const int m = lane & 15, half = lane >> 4, nn = lane & 15;
  f32x8 acc = {0.f, 0.f, 0.f, 0.f, 0.f, 0.f, 0.f, 0.f};
  for (int t = 0; t < 27; ++t) {
    const bf16_t* arow = dcnwp + (size_t)(t * 64 + o0 + m) * 64;
    const bf16_t* brow = &smp[t][nn][0];
    if (t < 26) __builtin_prefetch(dcnwp + (size_t)((t + 1) * 64 + o0 + m) * 64, 0, 1);
#pragma unroll
    for (int cb = 0; cb < 64; cb += 32)
      acc = wmma_bf16(frag_a(arow, cb, half), frag_b(brow, cb, half), acc);
  }
  bf16x8 sv;
#pragma unroll
  for (int r = 0; r < 8; ++r) sv[r] = (bf16_t)(acc[r] + dcnb[o0 + r + half * 8]);
  *(bf16x8*)(dcn_nl + (size_t)(n0 + nn) * 64 + o0 + half * 8) = sv;
}

// ---------------------------------------------------------------------------
// K6: conv1 GEMM, then h = u * a.  B: dcn_nl [N][64].  Out: h2nl [N][64] bf16.
// ---------------------------------------------------------------------------
__global__ void k_conv1_mul(const bf16_t* __restrict__ Wbf, const float* __restrict__ bias,
                            const bf16_t* __restrict__ Xnl, const float* __restrict__ u,
                            bf16_t* __restrict__ outnl) {
  __shared__ __align__(16) bf16_t Wl[64 * 64];
  const int tid = threadIdx.x;
  {
    const unsigned lbase = (unsigned)(uintptr_t)&Wl[0];
    const char* g = (const char*)Wbf;
    lds_async_b128(lbase + tid * 16u, g + tid * 16);
    lds_async_b128(lbase + 4096u + tid * 16u, g + 4096 + tid * 16);
    async_wait0();
  }
  __syncthreads();
  const int wave = tid >> 5, lane = tid & 31;
  const int ot = wave & 3, nt = wave >> 2;
  const int o0 = ot * 16;
  const int m = lane & 15, half = lane >> 4, nn = lane & 15;
  const int n = blockIdx.x * 32 + nt * 16 + nn;
  const bf16_t* arow = Wl + (o0 + m) * 64;
  const bf16_t* bcol = Xnl + (size_t)n * 64;
  f32x8 d = {0.f, 0.f, 0.f, 0.f, 0.f, 0.f, 0.f, 0.f};
#pragma unroll
  for (int cb = 0; cb < 64; cb += 32)
    d = wmma_bf16(frag_a(arow, cb, half), frag_b(bcol, cb, half), d);
  bf16x8 sv;
#pragma unroll
  for (int r = 0; r < 8; ++r) {
    const int o = o0 + r + half * 8;
    sv[r] = (bf16_t)((d[r] + bias[o]) * u[(size_t)o * NTOT + n]);
  }
  *(bf16x8*)(outnl + (size_t)n * 64 + o0 + half * 8) = sv;
}

// ---------------------------------------------------------------------------
// K7: proj2 GEMM + residual (x, already [N][64] f32) -> d_out [N][64] f32.
// ---------------------------------------------------------------------------
__global__ void k_proj2_out(const bf16_t* __restrict__ Wbf, const float* __restrict__ bias,
                            const bf16_t* __restrict__ Xnl, const float* __restrict__ xres,
                            float* __restrict__ out) {
  __shared__ __align__(16) bf16_t Wl[64 * 64];
  const int tid = threadIdx.x;
  {
    const unsigned lbase = (unsigned)(uintptr_t)&Wl[0];
    const char* g = (const char*)Wbf;
    lds_async_b128(lbase + tid * 16u, g + tid * 16);
    lds_async_b128(lbase + 4096u + tid * 16u, g + 4096 + tid * 16);
    async_wait0();
  }
  __syncthreads();
  const int wave = tid >> 5, lane = tid & 31;
  const int ot = wave & 3, nt = wave >> 2;
  const int o0 = ot * 16;
  const int m = lane & 15, half = lane >> 4, nn = lane & 15;
  const int n = blockIdx.x * 32 + nt * 16 + nn;
  const bf16_t* arow = Wl + (o0 + m) * 64;
  const bf16_t* bcol = Xnl + (size_t)n * 64;
  f32x8 d = {0.f, 0.f, 0.f, 0.f, 0.f, 0.f, 0.f, 0.f};
#pragma unroll
  for (int cb = 0; cb < 64; cb += 32)
    d = wmma_bf16(frag_a(arow, cb, half), frag_b(bcol, cb, half), d);
  const int obase = o0 + half * 8;
  const float4 r0 = *(const float4*)(xres + (size_t)n * 64 + obase);
  const float4 r1 = *(const float4*)(xres + (size_t)n * 64 + obase + 4);
  float4 s0, s1;
  s0.x = d[0] + bias[obase + 0] + r0.x;
  s0.y = d[1] + bias[obase + 1] + r0.y;
  s0.z = d[2] + bias[obase + 2] + r0.z;
  s0.w = d[3] + bias[obase + 3] + r0.w;
  s1.x = d[4] + bias[obase + 4] + r1.x;
  s1.y = d[5] + bias[obase + 5] + r1.y;
  s1.z = d[6] + bias[obase + 6] + r1.z;
  s1.w = d[7] + bias[obase + 7] + r1.w;
  *(float4*)(out + (size_t)n * 64 + obase) = s0;
  *(float4*)(out + (size_t)n * 64 + obase + 4) = s1;
}

// ---------------------------------------------------------------------------
extern "C" void kernel_launch(void* const* d_in, const int* in_sizes, int n_in,
                              void* d_out, int out_size, void* d_ws, size_t ws_size,
                              hipStream_t stream) {
  const float* x        = (const float*)d_in[0];
  const float* proj1_w  = (const float*)d_in[1];
  const float* proj1_b  = (const float*)d_in[2];
  const float* conv0_w  = (const float*)d_in[3];
  const float* conv0_b  = (const float*)d_in[4];
  const float* convsp_w = (const float*)d_in[5];
  const float* convsp_b = (const float*)d_in[6];
  const float* off_w    = (const float*)d_in[7];
  const float* off_b    = (const float*)d_in[8];
  const float* dcn_w    = (const float*)d_in[9];
  const float* dcn_b    = (const float*)d_in[10];
  const float* conv1_w  = (const float*)d_in[11];
  const float* conv1_b  = (const float*)d_in[12];
  const float* proj2_w  = (const float*)d_in[13];
  const float* proj2_b  = (const float*)d_in[14];
  float* out = (float*)d_out;

  char* ws = (char*)d_ws;
  size_t off = 0;
  auto carve = [&](size_t bytes) -> void* {
    void* p = ws + off;
    off += (bytes + 255) & ~(size_t)255;
    return p;
  };
  const size_t NC = (size_t)NTOT * CCH;
  bf16_t* xnl   = (bf16_t*)carve(NC * 2);                 // x cast to bf16 [N][64]
  float*  hbuf  = (float*)carve(NC * 4);                  // u [64][N] f32
  float*  a0    = (float*)carve(NC * 4);                  // dw5 out [64][N] f32
  bf16_t* a1nl  = (bf16_t*)carve(NC * 2);                 // dw7 out [N][64] bf16
  float*  offnl = (float*)carve((size_t)NTOT * 81 * 4);   // offsets [N][81] f32
  bf16_t* dcnnl = (bf16_t*)carve(NC * 2);                 // DCN out [N][64] bf16
  bf16_t* h2nl  = (bf16_t*)carve(NC * 2);                 // u*a [N][64] bf16
  bf16_t* w1    = (bf16_t*)carve(4096 * 2);
  bf16_t* w2    = (bf16_t*)carve(4096 * 2);
  bf16_t* w3    = (bf16_t*)carve(4096 * 2);
  bf16_t* offwp = (bf16_t*)carve((size_t)27 * 96 * 64 * 2);
  bf16_t* dcnwp = (bf16_t*)carve((size_t)27 * 64 * 64 * 2);
  (void)ws_size; (void)in_sizes; (void)n_in; (void)out_size;

  k_pack      <<<(27 * 96 * 64 + 255) / 256, 256, 0, stream>>>(
      proj1_w, conv1_w, proj2_w, off_w, dcn_w, w1, w2, w3, offwp, dcnwp);
  k_cast_bf16 <<<NC / (256 * 8), 256, 0, stream>>>(x, xnl);
  k_proj1_gelu<<<NTOT / 32, 256, 0, stream>>>(w1, proj1_b, xnl, hbuf);
  k_dw5       <<<dim3(NTOT / 256, CCH), 256, 0, stream>>>(hbuf, conv0_w, conv0_b, a0);
  k_dw7d3     <<<dim3(NTOT / 256, CCH), 256, 0, stream>>>(a0, convsp_w, convsp_b, a1nl);
  k_offconv   <<<NTOT / 16, 192, 0, stream>>>(offwp, off_b, a1nl, offnl);
  k_deform_dcn<<<NTOT / 16, 128, 0, stream>>>(offnl, a1nl, dcnwp, dcn_b, dcnnl);
  k_conv1_mul <<<NTOT / 32, 256, 0, stream>>>(w2, conv1_b, dcnnl, hbuf, h2nl);
  k_proj2_out <<<NTOT / 32, 256, 0, stream>>>(w3, proj2_b, h2nl, x, out);
}
